// DecoderViTBlock_1288490189565
// MI455X (gfx1250) — matrix operand order repaired and checked by reading, not compiled, not run on hardware
//
#include <hip/hip_runtime.h>
#include <math.h>

#define B_   2
#define N_   2048
#define C_   768
#define H_   12
#define HD_  64
#define HID_ 3072
#define TOK  (B_ * N_)        // 4096 rows
#define EPS_ 1e-5f
#define SCALE_ 0.125f         // 64^-0.5

typedef __attribute__((ext_vector_type(16))) __bf16 v16bf;
typedef __attribute__((ext_vector_type(8)))  float  v8f;
typedef __attribute__((__vector_size__(16))) int    i32x4;

union FragAB {
    v16bf v;
    unsigned short u[16];
    unsigned int   d[8];
};

__device__ __forceinline__ unsigned short f32_to_bf16(float f) {
    unsigned int u = __float_as_uint(f);
    unsigned int r = u + 0x7FFFu + ((u >> 16) & 1u);   // round-to-nearest-even
    return (unsigned short)(r >> 16);
}

// ---------------------------------------------------------------------------
// CDNA5 async global->LDS copy (ASYNCcnt-tracked), 16 bytes per lane.
// ---------------------------------------------------------------------------
#if __has_builtin(__builtin_amdgcn_global_load_async_to_lds_b128) && \
    __has_builtin(__builtin_amdgcn_s_wait_asynccnt)
#define HAS_ASYNC_LDS 1
#else
#define HAS_ASYNC_LDS 0
#endif

__device__ __forceinline__ void copy16_g2l(const unsigned short* gsrc,
                                           unsigned short* ldst) {
#if HAS_ASYNC_LDS
    __builtin_amdgcn_global_load_async_to_lds_b128(
        (__attribute__((address_space(1))) i32x4*)(unsigned long long)gsrc,
        (__attribute__((address_space(3))) i32x4*)(unsigned long long)ldst,
        0, 0);
#else
    *(uint4*)ldst = *(const uint4*)gsrc;
#endif
}
template <int CNT>
__device__ __forceinline__ void copy_wait_le() {
#if HAS_ASYNC_LDS
    __builtin_amdgcn_s_wait_asynccnt(CNT);
#endif
}

// ---------------------------------------------------------------------------
// 16-lane butterfly reductions on the VALU via DPP16 (no LDS traffic).
// Lanes 0-15 and 16-31 are separate DPP rows -> halves reduce independently,
// which matches the WMMA C-layout row mapping.
// ---------------------------------------------------------------------------
template <int CTRL>
__device__ __forceinline__ float dpp_movf(float x) {
    return __int_as_float(__builtin_amdgcn_update_dpp(
        0, __float_as_int(x), CTRL, 0xF, 0xF, true));
}
__device__ __forceinline__ float red16_max(float x) {
    x = fmaxf(x, dpp_movf<0xB1>(x));    // quad_perm [1,0,3,2]  (xor 1)
    x = fmaxf(x, dpp_movf<0x4E>(x));    // quad_perm [2,3,0,1]  (xor 2)
    x = fmaxf(x, dpp_movf<0x141>(x));   // row_half_mirror      (pairs 4-groups)
    x = fmaxf(x, dpp_movf<0x140>(x));   // row_mirror           (pairs 8-groups)
    return x;
}
__device__ __forceinline__ float red16_sum(float x) {
    x += dpp_movf<0xB1>(x);
    x += dpp_movf<0x4E>(x);
    x += dpp_movf<0x141>(x);
    x += dpp_movf<0x140>(x);
    return x;
}

// ---------------------------------------------------------------------------
// Weight convert + transpose: wt[n*K + k] = bf16(w[k*N + n])   (one-time)
// ---------------------------------------------------------------------------
__global__ __launch_bounds__(256) void wconv_kernel(
    const float* __restrict__ w, unsigned short* __restrict__ wt, int K, int N) {
    int idx = blockIdx.x * 256 + threadIdx.x;
    if (idx >= K * N) return;
    int k = idx / N, n = idx % N;
    wt[(size_t)n * K + k] = f32_to_bf16(w[idx]);
}

// ---------------------------------------------------------------------------
// LayerNorm: one 256-thread block per row; f32 in, bf16 out
// ---------------------------------------------------------------------------
__global__ __launch_bounds__(256) void ln_kernel(
    const float* __restrict__ x, const float* __restrict__ g,
    const float* __restrict__ b, unsigned short* __restrict__ out) {
    int row = blockIdx.x;
    const float* xr = x + (size_t)row * C_;
    unsigned short* orow = out + (size_t)row * C_;
    __shared__ float red[256];

    float s = 0.f;
    for (int c = threadIdx.x; c < C_; c += 256) s += xr[c];
    red[threadIdx.x] = s; __syncthreads();
    for (int o = 128; o > 0; o >>= 1) {
        if ((int)threadIdx.x < o) red[threadIdx.x] += red[threadIdx.x + o];
        __syncthreads();
    }
    float mu = red[0] * (1.f / C_);
    __syncthreads();

    float v = 0.f;
    for (int c = threadIdx.x; c < C_; c += 256) { float d = xr[c] - mu; v += d * d; }
    red[threadIdx.x] = v; __syncthreads();
    for (int o = 128; o > 0; o >>= 1) {
        if ((int)threadIdx.x < o) red[threadIdx.x] += red[threadIdx.x + o];
        __syncthreads();
    }
    float rstd = rsqrtf(red[0] * (1.f / C_) + EPS_);

    for (int c = threadIdx.x; c < C_; c += 256)
        orow[c] = f32_to_bf16((xr[c] - mu) * rstd * g[c] + b[c]);
}

// ---------------------------------------------------------------------------
// bf16 WMMA GEMM: out[M,N] = A[M,K] @ Bt[N,K]^T (+bias)(+GELU)(+residual)
// Block: 256 threads = 8 waves; BM=128, BN=128, BK=32; wave computes 16x128.
// Double-buffered async staging: 4 async instrs per wave per stage, so
// s_wait_asynccnt 4 after issuing the prefetch retires the current stage.
// ---------------------------------------------------------------------------
template <bool GELU, bool OUTBF16>
__global__ __launch_bounds__(256) void gemm_wmma(
    const unsigned short* __restrict__ A, const unsigned short* __restrict__ Bt,
    const float* __restrict__ bias, const float* __restrict__ resid,
    void* __restrict__ outp, int Ncols, int K) {
    __shared__ unsigned short Alds[2][128 * 32];   // [m][k]
    __shared__ unsigned short Blds[2][128 * 32];   // [n][k]

    const int wave = threadIdx.x >> 5;
    const int lane = threadIdx.x & 31;
    const int l16  = lane & 15;
    const int half = lane >> 4;
    const int m0 = blockIdx.y * 128;
    const int n0 = blockIdx.x * 128;

    auto stage = [&](int p, int k0) {
        #pragma unroll
        for (int i = 0; i < 2; ++i) {
            int idx = (int)threadIdx.x + i * 256;   // 0..511
            int r = idx >> 2, c = (idx & 3) * 8;    // 128 rows x 4 chunks
            copy16_g2l(A  + (size_t)(m0 + r) * K + k0 + c, &Alds[p][r * 32 + c]);
            copy16_g2l(Bt + (size_t)(n0 + r) * K + k0 + c, &Blds[p][r * 32 + c]);
        }
    };

    v8f acc[8] = {};
    stage(0, 0);
    int p = 0;
    for (int k0 = 0; k0 < K; k0 += 32, p ^= 1) {
        if (k0 + 32 < K) {
            stage(p ^ 1, k0 + 32);      // prefetch next K-slice
            copy_wait_le<4>();          // current stage (4 older instrs) done
        } else {
            copy_wait_le<0>();
        }
        __syncthreads();

        FragAB a;
        #pragma unroll
        for (int v = 0; v < 8; ++v) {
            int kk = 2 * v + 16 * half;
            a.d[v] = *(const unsigned int*)&Alds[p][(wave * 16 + l16) * 32 + kk];
        }
        #pragma unroll
        for (int t = 0; t < 8; ++t) {
            FragAB bf;
            int n = t * 16 + l16;
            #pragma unroll
            for (int v = 0; v < 8; ++v) {
                int kk = 2 * v + 16 * half;
                bf.d[v] = *(const unsigned int*)&Blds[p][n * 32 + kk];
            }
            acc[t] = __builtin_amdgcn_wmma_f32_16x16x32_bf16(
                false, a.v, false, bf.v, (short)0, acc[t], false, false);
        }
        __syncthreads();
    }

    float* outf = (float*)outp;
    unsigned short* outb = (unsigned short*)outp;
    #pragma unroll
    for (int t = 0; t < 8; ++t) {
        #pragma unroll
        for (int r = 0; r < 8; ++r) {
            int m = m0 + wave * 16 + r + 8 * half;
            int n = n0 + t * 16 + l16;
            float val = acc[t][r];
            if (bias) val += bias[n];
            if (GELU) val = 0.5f * val * (1.f + erff(val * 0.70710678118f));
            if (resid) val += resid[(size_t)m * Ncols + n];
            if (OUTBF16) outb[(size_t)m * Ncols + n] = f32_to_bf16(val);
            else         outf[(size_t)m * Ncols + n] = val;
        }
    }
}

// ---------------------------------------------------------------------------
// V transpose: vt[((b*H + h)*HD + hd)*N + n] = qkv[b,n][2C + h*HD + hd]
// ---------------------------------------------------------------------------
__global__ __launch_bounds__(256) void vtrans_kernel(
    const unsigned short* __restrict__ qkv, unsigned short* __restrict__ vt) {
    int idx = blockIdx.x * 256 + threadIdx.x;     // B*H*HD*N total
    int n  = idx & (N_ - 1);
    int r  = idx >> 11;                           // / N_
    int hd = r & (HD_ - 1);
    int bh = r >> 6;
    int b = bh / H_, h = bh % H_;
    vt[idx] = qkv[(size_t)(b * N_ + n) * (3 * C_) + 2 * C_ + h * HD_ + hd];
}

// ---------------------------------------------------------------------------
// Flash attention, bf16 in / bf16 out, double-buffered async K/V staging.
// qkv: [tok][3C] bf16; vt: [b,h,hd,n] bf16; y: [tok][C] bf16.
// Grid (N/64, B*H); 128 threads = 4 waves x 16 query rows; KV step = 32 keys.
// ---------------------------------------------------------------------------
__global__ __launch_bounds__(128) void attn_kernel(
    const unsigned short* __restrict__ qkv, const unsigned short* __restrict__ vt,
    unsigned short* __restrict__ y) {
    const int wave = threadIdx.x >> 5;
    const int lane = threadIdx.x & 31;
    const int l16  = lane & 15;
    const int half = lane >> 4;

    const int bh = blockIdx.y;
    const int b  = bh / H_;
    const int h  = bh % H_;
    const size_t tokbase = (size_t)b * N_;
    const int qrow = blockIdx.x * 64 + wave * 16;

    __shared__ unsigned short Klds[2][32 * 64];   // [key][hd]
    __shared__ unsigned short Vlds[2][64 * 32];   // [hd][key]
    __shared__ unsigned short Plds[4][16 * 32];   // per-wave P scratch [m][k]

    const unsigned short* kbase = qkv + tokbase * (size_t)(3 * C_) + C_ + h * HD_;
    const unsigned short* vbase = vt + (size_t)bh * HD_ * N_;

    auto stage = [&](int p, int j) {
        #pragma unroll
        for (int i = 0; i < 2; ++i) {
            int idx = (int)threadIdx.x + i * 128;             // 0..255
            int kkey = idx >> 3, kc = (idx & 7) * 8;
            copy16_g2l(kbase + (size_t)(j + kkey) * (3 * C_) + kc,
                       &Klds[p][kkey * 64 + kc]);
            int vhd = idx >> 2, vc = (idx & 3) * 8;
            copy16_g2l(vbase + (size_t)vhd * N_ + j + vc,
                       &Vlds[p][vhd * 32 + vc]);
        }
    };

    // Q fragments: bf16 pairs straight from global (once per wave)
    FragAB qf[2];
    #pragma unroll
    for (int s = 0; s < 2; ++s) {
        const unsigned short* qp =
            qkv + (tokbase + qrow + l16) * (size_t)(3 * C_) + h * HD_;
        #pragma unroll
        for (int v = 0; v < 8; ++v) {
            int hd = s * 32 + 2 * v + 16 * half;
            qf[s].d[v] = *(const unsigned int*)&qp[hd];
        }
    }

    v8f accO[4] = {};
    float mrow[8], lrow[8];
    #pragma unroll
    for (int r = 0; r < 8; ++r) { mrow[r] = -3.0e38f; lrow[r] = 0.f; }

    stage(0, 0);
    int p = 0;
    for (int j = 0; j < N_; j += 32, p ^= 1) {
        if (j + 32 < N_) {
            stage(p ^ 1, j + 32);
            copy_wait_le<4>();
        } else {
            copy_wait_le<0>();
        }
        __syncthreads();

        // S = Q @ K^T  (two 16-key tiles, K-dim 64 -> two wmma each)
        v8f S0 = {}, S1 = {};
        #pragma unroll
        for (int s = 0; s < 2; ++s) {
            FragAB b0, b1;
            #pragma unroll
            for (int v = 0; v < 8; ++v) {
                int hd = s * 32 + 2 * v + 16 * half;
                b0.d[v] = *(const unsigned int*)&Klds[p][l16 * 64 + hd];
                b1.d[v] = *(const unsigned int*)&Klds[p][(16 + l16) * 64 + hd];
            }
            S0 = __builtin_amdgcn_wmma_f32_16x16x32_bf16(
                false, qf[s].v, false, b0.v, (short)0, S0, false, false);
            S1 = __builtin_amdgcn_wmma_f32_16x16x32_bf16(
                false, qf[s].v, false, b1.v, (short)0, S1, false, false);
        }

        // Online softmax; row m = r + 8*half lives on (vgpr r, lane-half).
        // 16-lane reductions via DPP butterflies (VALU only).
        float p0[8], p1[8];
        #pragma unroll
        for (int r = 0; r < 8; ++r) {
            float s0 = S0[r] * SCALE_, s1 = S1[r] * SCALE_;
            float tm = red16_max(fmaxf(s0, s1));
            float mn = fmaxf(mrow[r], tm);
            float c  = __expf(mrow[r] - mn);
            p0[r] = __expf(s0 - mn);
            p1[r] = __expf(s1 - mn);
            float rs = red16_sum(p0[r] + p1[r]);
            lrow[r] = lrow[r] * c + rs;
            mrow[r] = mn;
            accO[0][r] *= c; accO[1][r] *= c; accO[2][r] *= c; accO[3][r] *= c;
        }

        // C-layout P -> A-layout via wave-private LDS scratch (same-wave LDS
        // ops are in-order; no barrier needed)
        #pragma unroll
        for (int r = 0; r < 8; ++r) {
            int m = r + 8 * half;
            Plds[wave][m * 32 + l16]      = f32_to_bf16(p0[r]);
            Plds[wave][m * 32 + 16 + l16] = f32_to_bf16(p1[r]);
        }
        FragAB pf;
        #pragma unroll
        for (int v = 0; v < 8; ++v) {
            int kk = 2 * v + 16 * half;
            pf.d[v] = *(const unsigned int*)&Plds[wave][l16 * 32 + kk];
        }

        // O += P @ V
        #pragma unroll
        for (int t = 0; t < 4; ++t) {
            FragAB vf;
            #pragma unroll
            for (int v = 0; v < 8; ++v) {
                int key = 2 * v + 16 * half;
                vf.d[v] = *(const unsigned int*)&Vlds[p][(t * 16 + l16) * 32 + key];
            }
            accO[t] = __builtin_amdgcn_wmma_f32_16x16x32_bf16(
                false, pf.v, false, vf.v, (short)0, accO[t], false, false);
        }
        __syncthreads();
    }

    #pragma unroll
    for (int t = 0; t < 4; ++t) {
        #pragma unroll
        for (int r = 0; r < 8; ++r) {
            int m  = qrow + r + 8 * half;
            int hd = t * 16 + l16;
            y[(tokbase + m) * (size_t)C_ + h * HD_ + hd] =
                f32_to_bf16(accO[t][r] / lrow[r]);
        }
    }
}

// ---------------------------------------------------------------------------
extern "C" void kernel_launch(void* const* d_in, const int* in_sizes, int n_in,
                              void* d_out, int out_size, void* d_ws, size_t ws_size,
                              hipStream_t stream) {
    (void)in_sizes; (void)n_in; (void)out_size; (void)ws_size;
    const float* x      = (const float*)d_in[0];
    const float* ln1_g  = (const float*)d_in[1];
    const float* ln1_b  = (const float*)d_in[2];
    const float* qkv_w  = (const float*)d_in[3];
    const float* proj_w = (const float*)d_in[4];
    const float* proj_b = (const float*)d_in[5];
    const float* ln2_g  = (const float*)d_in[6];
    const float* ln2_b  = (const float*)d_in[7];
    const float* fc1_w  = (const float*)d_in[8];
    const float* fc1_b  = (const float*)d_in[9];
    const float* fc2_w  = (const float*)d_in[10];
    const float* fc2_b  = (const float*)d_in[11];
    float* out = (float*)d_out;

    // workspace carve-up (all 256B-aligned by construction)
    char* p = (char*)d_ws;
    auto take = [&](size_t bytes) { char* q = p; p += (bytes + 255) & ~(size_t)255; return q; };
    unsigned short* wq_t  = (unsigned short*)take((size_t)3 * C_ * C_ * 2);   // [2304][768]
    unsigned short* wp_t  = (unsigned short*)take((size_t)C_ * C_ * 2);       // [768][768]
    unsigned short* w1_t  = (unsigned short*)take((size_t)HID_ * C_ * 2);     // [3072][768]
    unsigned short* w2_t  = (unsigned short*)take((size_t)C_ * HID_ * 2);     // [768][3072]
    unsigned short* hbuf  = (unsigned short*)take((size_t)TOK * C_ * 2);
    unsigned short* qkvb  = (unsigned short*)take((size_t)TOK * 3 * C_ * 2);
    unsigned short* vtb   = (unsigned short*)take((size_t)TOK * C_ * 2);
    unsigned short* ybuf  = (unsigned short*)take((size_t)TOK * C_ * 2);
    float*          x1buf = (float*)take((size_t)TOK * C_ * 4);
    unsigned short* mlpb  = (unsigned short*)take((size_t)TOK * HID_ * 2);

    dim3 blk256(256);

    // 0. one-time weight convert + transpose to bf16 [N][K]
    wconv_kernel<<<(C_ * 3 * C_ + 255) / 256, 256, 0, stream>>>(qkv_w, wq_t, C_, 3 * C_);
    wconv_kernel<<<(C_ * C_ + 255) / 256, 256, 0, stream>>>(proj_w, wp_t, C_, C_);
    wconv_kernel<<<(C_ * HID_ + 255) / 256, 256, 0, stream>>>(fc1_w, w1_t, C_, HID_);
    wconv_kernel<<<(HID_ * C_ + 255) / 256, 256, 0, stream>>>(fc2_w, w2_t, HID_, C_);

    // 1. LN1 -> bf16
    ln_kernel<<<TOK, 256, 0, stream>>>(x, ln1_g, ln1_b, hbuf);
    // 2. QKV = LN1(x) @ qkv_w -> bf16       [4096 x 2304 x 768]
    gemm_wmma<false, true><<<dim3(3 * C_ / 128, TOK / 128), blk256, 0, stream>>>(
        hbuf, wq_t, nullptr, nullptr, qkvb, 3 * C_, C_);
    // 3. V transpose -> [b,h,hd,n]
    vtrans_kernel<<<(TOK * C_) / 256, 256, 0, stream>>>(qkvb, vtb);
    // 4. attention -> y (bf16)
    attn_kernel<<<dim3(N_ / 64, B_ * H_), dim3(128), 0, stream>>>(qkvb, vtb, ybuf);
    // 5. x1 = x + y @ proj_w + proj_b (f32)
    gemm_wmma<false, false><<<dim3(C_ / 128, TOK / 128), blk256, 0, stream>>>(
        ybuf, wp_t, proj_b, x, x1buf, C_, C_);
    // 6. LN2 -> bf16
    ln_kernel<<<TOK, 256, 0, stream>>>(x1buf, ln2_g, ln2_b, hbuf);
    // 7. mlp = gelu(LN2(x1) @ fc1_w + fc1_b) -> bf16   [4096 x 3072 x 768]
    gemm_wmma<true, true><<<dim3(HID_ / 128, TOK / 128), blk256, 0, stream>>>(
        hbuf, w1_t, fc1_b, nullptr, mlpb, HID_, C_);
    // 8. out = x1 + mlp @ fc2_w + fc2_b (f32)           [4096 x 768 x 3072]
    gemm_wmma<false, false><<<dim3(C_ / 128, TOK / 128), blk256, 0, stream>>>(
        mlpb, w2_t, fc2_b, x1buf, out, C_, HID_);
}
